// mUnet_ARTfusion_SeqConcat_30923764531585
// MI455X (gfx1250) — matrix operand order repaired and checked by reading, not compiled, not run on hardware
//
#include <hip/hip_runtime.h>
#include <cmath>
#include <cstdint>

// ---------------------------------------------------------------------------
// Types for CDNA5 WMMA (wave32): A/B fragments are 16 bf16 (8 VGPRs),
// C/D accumulator is 8 f32 (8 VGPRs).
// ---------------------------------------------------------------------------
typedef __bf16 bf16_t;
typedef __bf16 bf16x16 __attribute__((ext_vector_type(16)));
typedef float  f32x8   __attribute__((ext_vector_type(8)));
typedef float  f32x4   __attribute__((ext_vector_type(4)));
typedef unsigned int u32x4 __attribute__((ext_vector_type(4)));

static inline int ceilDiv(int a, int b) { return (a + b - 1) / b; }
static inline dim3 gs(long n) {
  long b = (n + 255) / 256;
  if (b > 65535) b = 65535;
  if (b < 1) b = 1;
  return dim3((unsigned)b);
}

// ---------------------------------------------------------------------------
// bf16 WMMA GEMM:  C = alpha * A*B (+bias) with optional fused activation.
//   A: [M,K] f32 row-major (lda).  B: [K,N] f32 (ldb), or [N,K] if bT.
//   Two-level batching: z -> (b1 = z / batch2, b2 = z % batch2), with
//   independent element strides per level for A, B, C (attention heads).
// Block = 256 threads = 8 wave32 waves; block tile 128x64, K-step 32.
// Each wave: 32x32 C tile = 2x2 v_wmma_f32_16x16x32_bf16 accumulators.
// Staging: f32 global -> packed bf16 cvt in VGPRs -> b128 LDS stores.
// ---------------------------------------------------------------------------
__global__ __launch_bounds__(256)
void k_gemm_bf16_wmma(const float* __restrict__ A, const float* __restrict__ B,
                      float* __restrict__ C, const float* __restrict__ bias,
                      int M, int N, int K, int lda, int ldb, int ldc,
                      float alpha, int act, int bT, int vec, int batch2,
                      long sA1, long sA2, long sB1, long sB2, long sC1, long sC2)
{
  __shared__ bf16_t lA[128 * 32];  // [m][k]
  __shared__ bf16_t lBt[64 * 32];  // [n][k] (transposed for contiguous frag reads)

  const int z  = blockIdx.z;
  const int b1 = z / batch2;
  const int b2 = z - b1 * batch2;
  A += (long)b1 * sA1 + (long)b2 * sA2;
  B += (long)b1 * sB1 + (long)b2 * sB2;
  C += (long)b1 * sC1 + (long)b2 * sC2;

  const int row0 = blockIdx.y * 128;
  const int col0 = blockIdx.x * 64;
  const int tid  = threadIdx.x;
  const int lane = tid & 31;
  const int wv   = tid >> 5;          // wave 0..7
  const int wm   = (wv >> 1) * 32;    // wave row offset within block tile
  const int wn   = (wv & 1) * 32;     // wave col offset within block tile
  const int kh   = lane >> 4;         // half-wave select
  const int l15  = lane & 15;

  const bool fullM = (row0 + 128 <= M);
  const bool fullN = (col0 + 64 <= N);

  f32x8 acc[2][2];
#pragma unroll
  for (int i = 0; i < 2; ++i)
#pragma unroll
    for (int j = 0; j < 2; ++j) {
      f32x8 zero = {0.f, 0.f, 0.f, 0.f, 0.f, 0.f, 0.f, 0.f};
      acc[i][j] = zero;
    }

  for (int k0 = 0; k0 < K; k0 += 32) {
    const bool fullK = (k0 + 32 <= K);

    // ---- stage A tile (128x32), f32 -> bf16 ----
    if (vec && fullM && fullK) {
      // thread t: 16 contiguous K-elements of one row-half (b128 loads)
      const int i0 = tid * 16;
      const int r = i0 >> 5, kk = i0 & 31;
      const f32x4* pA = (const f32x4*)(A + (long)(row0 + r) * lda + k0 + kk);
      f32x4 f0 = pA[0], f1 = pA[1], f2 = pA[2], f3 = pA[3];
      union { bf16_t h[16]; u32x4 q[2]; } u;
#pragma unroll
      for (int e = 0; e < 4; ++e) {
        u.h[e]      = (bf16_t)f0[e];
        u.h[4 + e]  = (bf16_t)f1[e];
        u.h[8 + e]  = (bf16_t)f2[e];
        u.h[12 + e] = (bf16_t)f3[e];
      }
      *(u32x4*)&lA[i0]     = u.q[0];
      *(u32x4*)&lA[i0 + 8] = u.q[1];
    } else {
      // branch-free clamped loads + select
#pragma unroll 4
      for (int i = tid; i < 128 * 32; i += 256) {
        int r = i >> 5, k = i & 31;
        int gr = row0 + r, gk = k0 + k;
        int grc = gr < M ? gr : M - 1;
        int gkc = gk < K ? gk : K - 1;
        float v = A[(long)grc * lda + gkc];
        lA[i] = (bf16_t)((gr < M && gk < K) ? v : 0.f);
      }
    }

    // ---- stage B tile as [n][k] ----
    if (vec && bT && fullN && fullK) {
      const int i0 = tid * 8;
      const int n = i0 >> 5, kk = i0 & 31;
      const f32x4* pB = (const f32x4*)(B + (long)(col0 + n) * ldb + k0 + kk);
      f32x4 f0 = pB[0], f1 = pB[1];
      union { bf16_t h[8]; u32x4 q; } u;
#pragma unroll
      for (int e = 0; e < 4; ++e) { u.h[e] = (bf16_t)f0[e]; u.h[4 + e] = (bf16_t)f1[e]; }
      *(u32x4*)&lBt[i0] = u.q;
    } else if (vec && !bT && fullN && fullK) {
      // vector loads along n, transpose on the LDS store
      const int k = tid >> 3, n0 = (tid & 7) * 8;
      const f32x4* pB = (const f32x4*)(B + (long)(k0 + k) * ldb + col0 + n0);
      f32x4 f0 = pB[0], f1 = pB[1];
#pragma unroll
      for (int e = 0; e < 4; ++e) {
        lBt[(n0 + e) * 32 + k]     = (bf16_t)f0[e];
        lBt[(n0 + 4 + e) * 32 + k] = (bf16_t)f1[e];
      }
    } else {
#pragma unroll 4
      for (int i = tid; i < 64 * 32; i += 256) {
        int n = i >> 5, k = i & 31;
        int gn = col0 + n, gk = k0 + k;
        int gnc = gn < N ? gn : N - 1;
        int gkc = gk < K ? gk : K - 1;
        float v = bT ? B[(long)gnc * ldb + gkc] : B[(long)gkc * ldb + gnc];
        lBt[i] = (bf16_t)((gn < N && gk < K) ? v : 0.f);
      }
    }
    __syncthreads();

    if (k0 + 32 < K)  // prefetch next A tile -> global_prefetch_b8
      __builtin_prefetch(A + (long)(row0 + (tid >> 1)) * lda + k0 + 32, 0, 1);

    // Build fragments per ISA 16-bit A (16x32) / B (32x16) lane layouts.
    bf16x16 af[2], bfr[2];
#pragma unroll
    for (int t = 0; t < 2; ++t) {
      const bf16_t* pa = &lA[(wm + t * 16 + l15) * 32 + kh * 8];
#pragma unroll
      for (int e = 0; e < 8; ++e) { af[t][e] = pa[e]; af[t][e + 8] = pa[16 + e]; }
      const bf16_t* pb = &lBt[(wn + t * 16 + l15) * 32 + kh * 8];
#pragma unroll
      for (int e = 0; e < 8; ++e) { bfr[t][e] = pb[e]; bfr[t][e + 8] = pb[16 + e]; }
    }
#pragma unroll
    for (int mi = 0; mi < 2; ++mi)
#pragma unroll
      for (int ni = 0; ni < 2; ++ni)
        acc[mi][ni] = __builtin_amdgcn_wmma_f32_16x16x32_bf16(
            false, af[mi], false, bfr[ni], (short)0, acc[mi][ni], false, false);
    __syncthreads();
  }

  // Epilogue: C/D layout -> element i holds M = i + (lane<16 ? 0 : 8), N = lane&15
  const int baseR = row0 + wm + kh * 8;
  const int baseC = col0 + wn + l15;
  if (fullM && fullN) {
#pragma unroll
    for (int mi = 0; mi < 2; ++mi)
#pragma unroll
      for (int ni = 0; ni < 2; ++ni)
#pragma unroll
        for (int i = 0; i < 8; ++i) {
          int r  = baseR + mi * 16 + i;
          int cc = baseC + ni * 16;
          float v = acc[mi][ni][i] * alpha;
          if (bias) v += bias[cc];
          if (act == 1)      v = 0.5f * v * (1.f + erff(v * 0.70710678f));
          else if (act == 2) v = (v >= 0.f) ? v : 0.2f * v;
          C[(long)r * ldc + cc] = v;
        }
  } else {
#pragma unroll
    for (int mi = 0; mi < 2; ++mi)
#pragma unroll
      for (int ni = 0; ni < 2; ++ni)
#pragma unroll
        for (int i = 0; i < 8; ++i) {
          int r  = baseR + mi * 16 + i;
          int cc = baseC + ni * 16;
          if (r < M && cc < N) {
            float v = acc[mi][ni][i] * alpha;
            if (bias) v += bias[cc];
            if (act == 1)      v = 0.5f * v * (1.f + erff(v * 0.70710678f));
            else if (act == 2) v = (v >= 0.f) ? v : 0.2f * v;
            C[(long)r * ldc + cc] = v;
          }
        }
  }
}

// ---------------------------------------------------------------------------
// Supporting kernels (NHWC activation layout throughout)
// ---------------------------------------------------------------------------

// im2col for one image: x [H,W,C] -> col [H*W, C*9], k = c*9 + dy*3 + dx
__global__ void k_im2col_nhwc(const float* __restrict__ x, float* __restrict__ col,
                              int H, int W, int C)
{
  long total = (long)H * W * C * 9;
  for (long t = (long)blockIdx.x * blockDim.x + threadIdx.x; t < total;
       t += (long)gridDim.x * blockDim.x) {
    int k = (int)(t % (C * 9));
    long m = t / (C * 9);
    int c = k / 9; int r9 = k % 9;
    int dy = r9 / 3 - 1, dx = r9 % 3 - 1;
    int w = (int)(m % W); int h = (int)(m / W);
    int hh = h + dy, ww = w + dx;
    float v = 0.f;
    if (hh >= 0 && hh < H && ww >= 0 && ww < W) v = x[((long)hh * W + ww) * C + c];
    col[t] = v;
  }
}

// in-place instance norm + leaky relu; one block per (b,c)
__global__ __launch_bounds__(256)
void k_inorm_lrelu(float* __restrict__ x, int HW, int C)
{
  int c = blockIdx.x % C;
  long base = (long)(blockIdx.x / C) * HW * C + c;
  __shared__ float s1[256], s2[256];
  float sum = 0.f, sq = 0.f;
  for (int i = threadIdx.x; i < HW; i += 256) {
    float v = x[base + (long)i * C]; sum += v; sq += v * v;
  }
  s1[threadIdx.x] = sum; s2[threadIdx.x] = sq; __syncthreads();
  for (int s = 128; s > 0; s >>= 1) {
    if ((int)threadIdx.x < s) { s1[threadIdx.x] += s1[threadIdx.x + s]; s2[threadIdx.x] += s2[threadIdx.x + s]; }
    __syncthreads();
  }
  float m  = s1[0] / HW;
  float vr = s2[0] / HW - m * m;
  float rs = rsqrtf(vr + 1e-5f);
  for (int i = threadIdx.x; i < HW; i += 256) {
    long idx = base + (long)i * C;
    float v = (x[idx] - m) * rs;
    x[idx] = (v >= 0.f) ? v : 0.2f * v;
  }
}

// layernorm over last dim; one block per token row
__global__ __launch_bounds__(128)
void k_layernorm(const float* __restrict__ x, float* __restrict__ y,
                 const float* __restrict__ w, const float* __restrict__ b, int C)
{
  long base = (long)blockIdx.x * C;
  __shared__ float s1[128], s2[128];
  float sum = 0.f, sq = 0.f;
  for (int i = threadIdx.x; i < C; i += 128) { float v = x[base + i]; sum += v; sq += v * v; }
  s1[threadIdx.x] = sum; s2[threadIdx.x] = sq; __syncthreads();
  for (int s = 64; s > 0; s >>= 1) {
    if ((int)threadIdx.x < s) { s1[threadIdx.x] += s1[threadIdx.x + s]; s2[threadIdx.x] += s2[threadIdx.x + s]; }
    __syncthreads();
  }
  float m  = s1[0] / C;
  float vr = s2[0] / C - m * m;
  float rs = rsqrtf(vr + 1e-5f);
  for (int i = threadIdx.x; i < C; i += 128) y[base + i] = (x[base + i] - m) * rs * w[i] + b[i];
}

// row softmax (in place); one block per row of length N
__global__ __launch_bounds__(128)
void k_softmax(float* __restrict__ s, int N)
{
  long base = (long)blockIdx.x * N;
  __shared__ float red[128];
  float mx = -3.4e38f;
  for (int i = threadIdx.x; i < N; i += 128) mx = fmaxf(mx, s[base + i]);
  red[threadIdx.x] = mx; __syncthreads();
  for (int k = 64; k > 0; k >>= 1) {
    if ((int)threadIdx.x < k) red[threadIdx.x] = fmaxf(red[threadIdx.x], red[threadIdx.x + k]);
    __syncthreads();
  }
  mx = red[0]; __syncthreads();
  float sum = 0.f;
  for (int i = threadIdx.x; i < N; i += 128) { float e = __expf(s[base + i] - mx); s[base + i] = e; sum += e; }
  red[threadIdx.x] = sum; __syncthreads();
  for (int k = 64; k > 0; k >>= 1) {
    if ((int)threadIdx.x < k) red[threadIdx.x] += red[threadIdx.x + k];
    __syncthreads();
  }
  float inv = 1.f / red[0];
  for (int i = threadIdx.x; i < N; i += 128) s[base + i] *= inv;
}

// window / interleave partition: x [B,H,W,C] -> y [Bn, dstN, C] at row offset noff
__global__ void k_partition(const float* __restrict__ x, float* __restrict__ y,
                            int B, int H, int W, int C, int G, int I, int ds,
                            int dstN, int noff)
{
  long total = (long)B * H * W * C;
  for (long t = (long)blockIdx.x * blockDim.x + threadIdx.x; t < total;
       t += (long)gridDim.x * blockDim.x) {
    int c = (int)(t % C);
    long r = t / C;
    int n = (int)(r % 100);
    long bn = r / 100;
    int b, h, w;
    if (ds == 0) {
      int WG = W / G, HG = H / G;
      int gw = (int)(bn % WG);
      int gh = (int)((bn / WG) % HG);
      b = (int)(bn / ((long)WG * HG));
      h = gh * G + n / G; w = gw * G + n % G;
    } else {
      int jj = (int)(bn % I), ii = (int)((bn / I) % I);
      b = (int)(bn / ((long)I * I));
      int WI = W / I;
      h = (n / WI) * I + ii; w = (n % WI) * I + jj;
    }
    y[(bn * dstN + noff + n) * C + c] = x[(((long)b * H + h) * W + w) * C + c];
  }
}

// inverse partition with residual add: x += unpartition(y)
__global__ void k_unpartition_add(float* __restrict__ x, const float* __restrict__ y,
                                  int B, int H, int W, int C, int G, int I, int ds,
                                  int srcN, int noff)
{
  long total = (long)B * H * W * C;
  for (long t = (long)blockIdx.x * blockDim.x + threadIdx.x; t < total;
       t += (long)gridDim.x * blockDim.x) {
    int c = (int)(t % C);
    long r = t / C;
    int n = (int)(r % 100);
    long bn = r / 100;
    int b, h, w;
    if (ds == 0) {
      int WG = W / G, HG = H / G;
      int gw = (int)(bn % WG);
      int gh = (int)((bn / WG) % HG);
      b = (int)(bn / ((long)WG * HG));
      h = gh * G + n / G; w = gw * G + n % G;
    } else {
      int jj = (int)(bn % I), ii = (int)((bn / I) % I);
      b = (int)(bn / ((long)I * I));
      int WI = W / I;
      h = (n / WI) * I + ii; w = (n % WI) * I + jj;
    }
    x[(((long)b * H + h) * W + w) * C + c] += y[(bn * srcN + noff + n) * C + c];
  }
}

// 2x2 average pool, NHWC
__global__ void k_avgpool(const float* __restrict__ x, float* __restrict__ y,
                          int B, int H, int W, int C)
{
  int Ho = H / 2, Wo = W / 2;
  long total = (long)B * Ho * Wo * C;
  for (long t = (long)blockIdx.x * blockDim.x + threadIdx.x; t < total;
       t += (long)gridDim.x * blockDim.x) {
    int c = (int)(t % C);
    long r = t / C;
    int w = (int)(r % Wo); r /= Wo;
    int h = (int)(r % Ho);
    int b = (int)(r / Ho);
    const float* p = x + (((long)b * H + 2 * h) * W + 2 * w) * C + c;
    y[t] = 0.25f * (p[0] + p[C] + p[(long)W * C] + p[(long)W * C + C]);
  }
}

// generic strided column copy (concat / split along channel dim)
__global__ void k_copy_cols(const float* __restrict__ src, int lds_, int soff,
                            float* __restrict__ dst, int ldd, int doff,
                            long T, int Cn)
{
  long total = T * Cn;
  for (long t = (long)blockIdx.x * blockDim.x + threadIdx.x; t < total;
       t += (long)gridDim.x * blockDim.x) {
    int c = (int)(t % Cn);
    long r = t / Cn;
    dst[r * ldd + doff + c] = src[r * lds_ + soff + c];
  }
}

__global__ void k_add(float* __restrict__ a, const float* __restrict__ b, long n)
{
  for (long t = (long)blockIdx.x * blockDim.x + threadIdx.x; t < n;
       t += (long)gridDim.x * blockDim.x) a[t] += b[t];
}

// scatter GEMM result [B*H*W, Co*4] -> NHWC [B,2H,2W,Co] (2x2 transposed conv)
__global__ void k_scatter_tconv(const float* __restrict__ g, float* __restrict__ y,
                                int B, int H, int W, int Co)
{
  long total = (long)B * H * W * Co * 4;
  for (long t = (long)blockIdx.x * blockDim.x + threadIdx.x; t < total;
       t += (long)gridDim.x * blockDim.x) {
    int q = (int)(t % 4);
    int o = (int)((t / 4) % Co);
    long m = t / (4 * Co);
    int w = (int)(m % W);
    long r = m / W;
    int h = (int)(r % H);
    int b = (int)(r / H);
    int rr = q >> 1, ss = q & 1;
    y[(((long)b * 2 * H + 2 * h + rr) * (2 * W) + 2 * w + ss) * (long)Co + o] = g[t];
  }
}

// ---------------------------------------------------------------------------
// Host-side orchestration
// ---------------------------------------------------------------------------
struct Arena {
  char* base; size_t cap; size_t off;
  float* f(long n) {
    size_t b = (off + 255) & ~(size_t)255;
    off = b + (size_t)n * 4;
    return (float*)(base + b);
  }
};

struct Ctx { hipStream_t st; Arena* ar; float* col; };

static void gemm(hipStream_t st, const float* A, const float* B, float* C,
                 const float* bias, int M, int N, int K, int lda, int ldb, int ldc,
                 float alpha, int act, int bT,
                 int nb = 1, int batch2 = 1,
                 long sA1 = 0, long sA2 = 0, long sB1 = 0, long sB2 = 0,
                 long sC1 = 0, long sC2 = 0)
{
  int vec = ((lda & 3) == 0 && (ldb & 3) == 0) ? 1 : 0;
  dim3 g((unsigned)ceilDiv(N, 64), (unsigned)ceilDiv(M, 128), (unsigned)nb);
  k_gemm_bf16_wmma<<<g, 256, 0, st>>>(A, B, C, bias, M, N, K, lda, ldb, ldc,
                                      alpha, act, bT, vec, batch2,
                                      sA1, sA2, sB1, sB2, sC1, sC2);
}

// one 3x3 conv (implicit GEMM, per-image im2col chunking to bound scratch)
static void conv3x3(Ctx& c, const float* x, const float* w, float* out,
                    int B, int H, int W, int Cin, int Cout)
{
  long Mimg = (long)H * W;
  int K = Cin * 9;
  for (int b = 0; b < B; ++b) {
    k_im2col_nhwc<<<gs(Mimg * K), 256, 0, c.st>>>(x + (long)b * Mimg * Cin, c.col, H, W, Cin);
    gemm(c.st, c.col, w, out + (long)b * Mimg * Cout, nullptr,
         (int)Mimg, Cout, K, K, K, Cout, 1.f, 0, /*bT=*/1);
  }
}

static float* convBlock(Ctx& c, const float* x, const float* w1, const float* w2,
                        int B, int H, int W, int Cin, int Cout)
{
  float* out = c.ar->f((long)B * H * W * Cout);
  size_t mark = c.ar->off;
  float* t1 = c.ar->f((long)B * H * W * Cout);
  conv3x3(c, x, w1, t1, B, H, W, Cin, Cout);
  k_inorm_lrelu<<<(unsigned)(B * Cout), 256, 0, c.st>>>(t1, H * W, Cout);
  conv3x3(c, t1, w2, out, B, H, W, Cout, Cout);
  k_inorm_lrelu<<<(unsigned)(B * Cout), 256, 0, c.st>>>(out, H * W, Cout);
  c.ar->off = mark;
  return out;
}

struct TB { const float *ln1w,*ln1b,*wqkv,*bqkv,*wo,*bo,*ln2w,*ln2b,*w1,*b1,*w2,*b2; };

static TB makeTB(void* const* d_in, int base)
{
  auto P = [&](int i) { return (const float*)d_in[i]; };
  TB t;
  t.b1   = P(base + 0);  t.b2   = P(base + 1);
  t.bo   = P(base + 2);  t.bqkv = P(base + 3);
  t.ln1b = P(base + 4);  t.ln1w = P(base + 5);
  t.ln2b = P(base + 6);  t.ln2w = P(base + 7);
  t.w1   = P(base + 8);  t.w2   = P(base + 9);
  t.wo   = P(base + 10); t.wqkv = P(base + 11);
  return t;
}

// windowed MHA over part [Bn, Ntok, C]; projected output -> attout [Bn, Ntok, C]
static void attention(Ctx& c, const TB& p, const float* part, float* attout,
                      long Bn, int Ntok, int C, int nh)
{
  size_t mark = c.ar->off;
  long Mq = Bn * Ntok;
  float* qkv = c.ar->f(Mq * 3 * C);
  gemm(c.st, part, p.wqkv, qkv, p.bqkv, (int)Mq, 3 * C, C, C, 3 * C, 3 * C, 1.f, 0, 0);

  int d = C / nh;
  float scale = 1.0f / sqrtf((float)d);
  long s3C = (long)Ntok * 3 * C;
  long sSc = (long)Ntok * Ntok;

  float* sc = c.ar->f(Bn * (long)nh * sSc);
  // scores = scale * q @ k^T  (batched over (Bn, nh))
  gemm(c.st, qkv, qkv + C, sc, nullptr, Ntok, Ntok, d, 3 * C, 3 * C, Ntok,
       scale, 0, /*bT=*/1, (int)(Bn * nh), nh, s3C, d, s3C, d, (long)nh * sSc, sSc);
  k_softmax<<<(unsigned)(Bn * nh * Ntok), 128, 0, c.st>>>(sc, Ntok);

  float* oatt = c.ar->f(Mq * C);
  // o = scores @ v
  gemm(c.st, sc, qkv + 2 * C, oatt, nullptr, Ntok, d, Ntok, Ntok, 3 * C, C,
       1.f, 0, 0, (int)(Bn * nh), nh, (long)nh * sSc, sSc, s3C, d, (long)Ntok * C, d);

  // projection
  gemm(c.st, oatt, p.wo, attout, p.bo, (int)Mq, C, C, C, C, C, 1.f, 0, 0);
  c.ar->off = mark;
}

static void mlpResidual(Ctx& c, const TB& p, float* x, long T, int C)
{
  float* ln = c.ar->f(T * C);
  k_layernorm<<<(unsigned)T, 128, 0, c.st>>>(x, ln, p.ln2w, p.ln2b, C);
  float* h1 = c.ar->f(T * 4 * C);
  gemm(c.st, ln, p.w1, h1, p.b1, (int)T, 4 * C, C, C, 4 * C, 4 * C, 1.f, /*gelu*/1, 0);
  float* h2 = c.ar->f(T * C);
  gemm(c.st, h1, p.w2, h2, p.b2, (int)T, C, 4 * C, 4 * C, C, C, 1.f, 0, 0);
  k_add<<<gs(T * C), 256, 0, c.st>>>(x, h2, T * C);
}

static void tblock(Ctx& c, const TB& p, float* x, int B, int H, int W, int C,
                   int nh, int G, int I, int ds)
{
  long T = (long)B * H * W;
  size_t mark = c.ar->off;
  float* ln = c.ar->f(T * C);
  k_layernorm<<<(unsigned)T, 128, 0, c.st>>>(x, ln, p.ln1w, p.ln1b, C);
  long Bn = T / 100;
  float* part = c.ar->f(T * C);
  k_partition<<<gs(T * C), 256, 0, c.st>>>(ln, part, B, H, W, C, G, I, ds, 100, 0);
  float* att = c.ar->f(T * C);
  attention(c, p, part, att, Bn, 100, C, nh);
  k_unpartition_add<<<gs(T * C), 256, 0, c.st>>>(x, att, B, H, W, C, G, I, ds, 100, 0);
  mlpResidual(c, p, x, T, C);
  c.ar->off = mark;
}

static void concatTBlock(Ctx& c, const TB& p, float* x1, float* x2,
                         int B, int H, int W, int C, int nh, int G, int I, int ds)
{
  long T = (long)B * H * W;
  long Bn = T / 100;
  size_t mark = c.ar->off;
  float* ln   = c.ar->f(T * C);
  float* part = c.ar->f(2 * T * C);
  k_layernorm<<<(unsigned)T, 128, 0, c.st>>>(x1, ln, p.ln1w, p.ln1b, C);
  k_partition<<<gs(T * C), 256, 0, c.st>>>(ln, part, B, H, W, C, G, I, ds, 200, 0);
  k_layernorm<<<(unsigned)T, 128, 0, c.st>>>(x2, ln, p.ln1w, p.ln1b, C);
  k_partition<<<gs(T * C), 256, 0, c.st>>>(ln, part, B, H, W, C, G, I, ds, 200, 100);

  float* att = c.ar->f(2 * T * C);
  attention(c, p, part, att, Bn, 200, C, nh);
  k_unpartition_add<<<gs(T * C), 256, 0, c.st>>>(x1, att, B, H, W, C, G, I, ds, 200, 0);
  k_unpartition_add<<<gs(T * C), 256, 0, c.st>>>(x2, att, B, H, W, C, G, I, ds, 200, 100);

  mlpResidual(c, p, x1, T, C);
  mlpResidual(c, p, x2, T, C);
  c.ar->off = mark;
}

static void runDecoder(Ctx& c, void* const* d_in, float* x, float* const* stk,
                       const int* cw1, const int* cw2, const int* up,
                       int fbIdx, int fwIdx, float* out, int B)
{
  auto P = [&](int i) { return (const float*)d_in[i]; };
  int H = 15, Cin = 512;
  for (int i = 0; i < 4; ++i) {
    int Co = Cin / 2;
    long M = (long)B * H * H;
    // transposed conv 2x2: GEMM [M,Cin] x [Cin, Co*4], then scatter
    float* g = c.ar->f(M * Co * 4);
    gemm(c.st, x, P(up[i]), g, nullptr, (int)M, Co * 4, Cin, Cin, Co * 4, Co * 4, 1.f, 0, 0);
    int H2 = 2 * H;
    long T2 = (long)B * H2 * H2;
    float* t = c.ar->f(T2 * Co);
    k_scatter_tconv<<<gs(M * Co * 4), 256, 0, c.st>>>(g, t, B, H, H, Co);
    k_inorm_lrelu<<<(unsigned)(B * Co), 256, 0, c.st>>>(t, H2 * H2, Co);
    // concat [up, skip] along channels
    float* u = c.ar->f(T2 * 2 * Co);
    k_copy_cols<<<gs(T2 * Co), 256, 0, c.st>>>(t, Co, 0, u, 2 * Co, 0, T2, Co);
    k_copy_cols<<<gs(T2 * Co), 256, 0, c.st>>>(stk[3 - i], Co, 0, u, 2 * Co, Co, T2, Co);
    x = convBlock(c, u, P(cw1[i]), P(cw2[i]), B, H2, H2, 2 * Co, Co);
    H = H2; Cin = Co;
  }
  // final 1x1 conv: [B*240*240, 32] x [32,1] + bias -> out
  gemm(c.st, x, P(fwIdx), out, P(fbIdx), B * 240 * 240, 1, 32, 32, 32, 1, 1.f, 0, /*bT=*/1);
}

// ---------------------------------------------------------------------------
// Entry point
// ---------------------------------------------------------------------------
extern "C" void kernel_launch(void* const* d_in, const int* in_sizes, int n_in,
                              void* d_out, int out_size, void* d_ws, size_t ws_size,
                              hipStream_t stream)
{
  (void)in_sizes; (void)n_in; (void)out_size;
  auto P = [&](int i) { return (const float*)d_in[i]; };

  // Flattened-params index map (sorted dict keys, lists in order):
  // 0:image1 1:image2 | 2-3:conv1.w1/w2 4-5:conv2 | dec1: 6..19 | dec2: 20..33
  // enc1: 34..41 | enc2: 42..49 | fuse tblocks: 50..241 (12 tensors each)
  static const int E1W1[4] = {34, 36, 38, 40}, E1W2[4] = {35, 37, 39, 41};
  static const int E2W1[4] = {42, 44, 46, 48}, E2W2[4] = {43, 45, 47, 49};
  static const int D1CW1[4] = {6, 9, 12, 15},   D1CW2[4] = {7, 10, 13, 16},   D1UP[4] = {8, 11, 14, 19};
  static const int D2CW1[4] = {20, 23, 26, 29}, D2CW2[4] = {21, 24, 27, 30},  D2UP[4] = {22, 25, 28, 33};
  const int D1FB = 17, D1FW = 18, D2FB = 31, D2FW = 32;
  static const int FUSE_BASE[4] = {50, 50 + 2 * 12, 50 + 6 * 12, 50 + 10 * 12};

  static const int Hs[4]    = {240, 120, 60, 30};
  static const int Ce[4]    = {32, 64, 128, 256};
  static const int HEADS[4] = {1, 2, 4, 8};
  static const int NB[4]    = {2, 4, 4, 6};
  static const int IV[4]    = {24, 12, 6, 3};
  const int B = 4, G = 10;

  Arena ar{(char*)d_ws, ws_size, 0};
  Ctx c{stream, &ar, nullptr};
  // im2col scratch (per-image), sized for worst case M=57600, K=576
  c.col = ar.f((long)57600 * 576);

  float* s1[4]; float* s2[4];
  const float* src1 = P(0);
  const float* src2 = P(1);
  int Cin = 1;

  for (int l = 0; l < 4; ++l) {
    int H = Hs[l], C = Ce[l];
    float* a1 = convBlock(c, src1, P(E1W1[l]), P(E1W2[l]), B, H, H, Cin, C);
    float* a2 = convBlock(c, src2, P(E2W1[l]), P(E2W2[l]), B, H, H, Cin, C);
    long T = (long)B * H * H;
    if (l < 2) {
      int C2 = 2 * C;
      float* xc = ar.f(T * C2);
      k_copy_cols<<<gs(T * C), 256, 0, stream>>>(a1, C, 0, xc, C2, 0, T, C);
      k_copy_cols<<<gs(T * C), 256, 0, stream>>>(a2, C, 0, xc, C2, C, T, C);
      for (int i = 0; i < NB[l]; ++i) {
        TB tb = makeTB(d_in, FUSE_BASE[l] + 12 * i);
        tblock(c, tb, xc, B, H, H, C2, HEADS[l], G, IV[l], i % 2);
      }
      k_copy_cols<<<gs(T * C), 256, 0, stream>>>(xc, C2, 0, a1, C, 0, T, C);
      k_copy_cols<<<gs(T * C), 256, 0, stream>>>(xc, C2, C, a2, C, 0, T, C);
    } else {
      for (int i = 0; i < NB[l]; ++i) {
        TB tb = makeTB(d_in, FUSE_BASE[l] + 12 * i);
        concatTBlock(c, tb, a1, a2, B, H, H, C, HEADS[l], G, IV[l], i % 2);
      }
    }
    s1[l] = a1; s2[l] = a2;
    int Ho = H / 2;
    float* p1 = ar.f((long)B * Ho * Ho * C);
    float* p2 = ar.f((long)B * Ho * Ho * C);
    k_avgpool<<<gs((long)B * Ho * Ho * C), 256, 0, stream>>>(a1, p1, B, H, H, C);
    k_avgpool<<<gs((long)B * Ho * Ho * C), 256, 0, stream>>>(a2, p2, B, H, H, C);
    src1 = p1; src2 = p2; Cin = C;
  }

  // bottleneck: 256 -> 512 at 15x15
  float* bt1 = convBlock(c, src1, P(2), P(3), B, 15, 15, 256, 512);
  float* bt2 = convBlock(c, src2, P(4), P(5), B, 15, 15, 256, 512);

  float* out1 = (float*)d_out;
  float* out2 = out1 + (long)B * 240 * 240;
  runDecoder(c, d_in, bt1, s1, D1CW1, D1CW2, D1UP, D1FB, D1FW, out1, B);
  runDecoder(c, d_in, bt2, s2, D2CW1, D2CW2, D2UP, D2FB, D2FW, out2, B);
}
